// CAB_52390011077355
// MI455X (gfx1250) — matrix operand order repaired and checked by reading, not compiled, not run on hardware
//
#include <hip/hip_runtime.h>
#include <hip/hip_bf16.h>

// ---------------------------------------------------------------------------
// Channel-attention block (Restormer-style) for MI455X / gfx1250.
// Memory-bound (~10 GFLOP vs ~470MB @ 23.3 TB/s => ~20us floor):
//   - f16 intermediates, all GEMMs on v_wmma_f32_16x16x32_f16
//   - proj folded into attn@v GEMM (one fewer HBM round trip)
//   - activation tiles staged to LDS with gfx1250 async global->LDS DMA
//     (ASYNCcnt) when the builtin is available.
// ---------------------------------------------------------------------------

typedef __attribute__((ext_vector_type(16))) _Float16 v16h;
typedef __attribute__((ext_vector_type(8)))  float    v8f;
typedef int vi4 __attribute__((__vector_size__(16))); // native 16B vector

#define NPIX 65536   // H*W = 256*256
#define NB   4       // batch
#define PITCH 72     // LDS row pitch (elements): 16B-aligned rows for f16/f32

#if defined(__HIP_DEVICE_COMPILE__) && defined(__has_builtin)
#if __has_builtin(__builtin_amdgcn_global_load_async_to_lds_b128) && \
    __has_builtin(__builtin_amdgcn_s_wait_asynccnt)
#define CAB_HAVE_ASYNC_LDS 1
#endif
#endif

// --------------------------- zero small accumulators -----------------------
__global__ void CAB_zero_kernel(float* p, int n) {
  int i = blockIdx.x * 256 + threadIdx.x;
  if (i < n) p[i] = 0.0f;
}

// --------------------------- 1x1-conv GEMM via WMMA ------------------------
// Y[b][co0+M][p] = sum_ci W[b?][co0+M][ci] * X[b][ci][p],  Cin = 64 fixed.
// Block: 256 thr = 8 waves; block tile 64 rows x 64 pixels; wave = 16x32.
template <typename TS>
__global__ void CAB_gemm_kernel(const TS* __restrict__ X,
                                const float* __restrict__ W,
                                float* __restrict__ Y,
                                long xbs, long ybs, int wbs) {
  const int tid  = threadIdx.x;
  const int lane = tid & 31;
  const int w    = tid >> 5;
  const int wm   = w >> 1;          // 0..3 -> 16-row tile
  const int wn   = w & 1;           // 0..1 -> 32-pixel half
  const int col  = lane & 15;
  const int kh   = lane >> 4;       // lane half selects K sub-block
  const int b    = blockIdx.y;
  const int p0   = blockIdx.x * 64;
  const int co0  = blockIdx.z * 64;

  __shared__ TS xs[64 * PITCH];     // 64 ci x 64 px tile (raw source type)

  // ---- stage activation tile global -> LDS in 16B chunks ----
  const TS* xb = X + (long)b * xbs + p0;
  __builtin_prefetch(xb + 64, 0, 1);                 // global_prefetch_b8
  constexpr int EPC = 16 / (int)sizeof(TS);          // elements per 16B chunk
  constexpr int CPR = 64 / EPC;                      // chunks per ci-row
#pragma unroll
  for (int i = 0; i < (64 * CPR) / 256; ++i) {
    int chunk = tid + i * 256;
    int ci = chunk / CPR, cc = chunk % CPR;
    const TS* g = xb + (long)ci * NPIX + cc * EPC;
    TS*       l = &xs[ci * PITCH + cc * EPC];
#ifdef CAB_HAVE_ASYNC_LDS
    __builtin_amdgcn_global_load_async_to_lds_b128((vi4*)g, (vi4*)l, 0, 0);
#else
    *(vi4*)l = *(const vi4*)g;
#endif
  }

  // ---- A fragments: weights, row = co0 + 16*wm + (lane&15); K=64 => 2 steps
  const float* wrow = W + (long)b * wbs + (long)(co0 + 16 * wm + col) * 64;
  v16h a0, a1;
#pragma unroll
  for (int e = 0; e < 16; ++e) {
    int k0 = kh * 16 + e;                                  // contiguous K layout (matches B)
    a0[e] = (_Float16)wrow[k0];
    a1[e] = (_Float16)wrow[32 + k0];
  }
#ifdef CAB_HAVE_ASYNC_LDS
  __builtin_amdgcn_s_wait_asynccnt(0);
#endif
  __syncthreads();

  v8f acc0 = {}, acc1 = {};
#pragma unroll
  for (int ks = 0; ks < 2; ++ks) {
    v16h bf0, bf1;
#pragma unroll
    for (int e = 0; e < 16; ++e) {
      int K = ks * 32 + kh * 16 + e;                      // contiguous K layout (matches A)
      bf0[e] = (_Float16)xs[K * PITCH + wn * 32 + col];
      bf1[e] = (_Float16)xs[K * PITCH + wn * 32 + 16 + col];
    }
    v16h a = ks ? a1 : a0;
    acc0 = __builtin_amdgcn_wmma_f32_16x16x32_f16(false, a, false, bf0,
                                                  (short)0, acc0, false, false);
    acc1 = __builtin_amdgcn_wmma_f32_16x16x32_f16(false, a, false, bf1,
                                                  (short)0, acc1, false, false);
  }

  float* yb = Y + (long)b * ybs + (long)(co0 + 16 * wm) * NPIX + p0 + wn * 32;
#pragma unroll
  for (int r = 0; r < 8; ++r) {
    int M = r + 8 * kh;                                    // ISA C/D layout
    yb[(long)M * NPIX + col]      = acc0[r];
    yb[(long)M * NPIX + 16 + col] = acc1[r];
  }
}

// --------------------- depthwise 3x3 + f16 cast + sum-of-squares -----------
__global__ void CAB_dw_kernel(const float* __restrict__ T,
                              const float* __restrict__ DW,
                              _Float16* __restrict__ O,
                              float* __restrict__ ssq,
                              int ctotal, int ssq_limit) {
  const int c = blockIdx.y, b = blockIdx.z;
  const int p = blockIdx.x * 256 + threadIdx.x;
  const int y = p >> 8, x = p & 255;
  const float* tb = T + ((long)b * ctotal + c) * (long)NPIX;
  const float* wk = DW + c * 9;
  float s = 0.0f;
#pragma unroll
  for (int ky = 0; ky < 3; ++ky) {
    int yy = y + ky - 1;
    if (yy < 0 || yy > 255) continue;
#pragma unroll
    for (int kx = 0; kx < 3; ++kx) {
      int xx = x + kx - 1;
      if (xx < 0 || xx > 255) continue;
      s += wk[ky * 3 + kx] * tb[yy * 256 + xx];
    }
  }
  O[((long)b * ctotal + c) * (long)NPIX + p] = (_Float16)s;

  if (c < ssq_limit) {            // block-uniform branch (c is blockIdx.y)
    __shared__ float red[256];
    red[threadIdx.x] = s * s;
    __syncthreads();
    for (int off = 128; off > 0; off >>= 1) {
      if (threadIdx.x < off) red[threadIdx.x] += red[threadIdx.x + off];
      __syncthreads();
    }
    if (threadIdx.x == 0) atomicAdd(&ssq[b * 64 + c], red[0]);
  }
}

// ------------- Gram accumulation: attn_raw[c,d] = sum_p q[c,p]*k[d,p] ------
// Per (b, head-pair) a 16x16 WMMA accumulator; reduction over pixels split
// across 8 waves x 4 chunk-blocks, finished with f32 atomics.
__global__ void CAB_attn_kernel(const _Float16* __restrict__ Q,
                                const _Float16* __restrict__ Kp,
                                float* __restrict__ attnacc) {
  const int b = blockIdx.z, hp = blockIdx.y, chunk = blockIdx.x;
  const int tid = threadIdx.x, lane = tid & 31, w = tid >> 5;
  const int col = lane & 15, kh = lane >> 4;
  const int c0 = hp * 16;
  const long pix0 = ((long)chunk * 8 + w) * 2048;

  const _Float16* qrow = Q  + ((long)b * 64  + c0 + col) * (long)NPIX; // A row/lane
  const _Float16* krow = Kp + ((long)b * 128 + c0 + col) * (long)NPIX; // B col/lane

  v8f acc = {};
  for (int it = 0; it < 64; ++it) {
    long pb = pix0 + it * 32;
    v16h af, bf;
#pragma unroll
    for (int e = 0; e < 16; ++e) {
      int ka = kh * 16 + e;
      af[e] = qrow[pb + ka];
      bf[e] = krow[pb + kh * 16 + e];
    }
    acc = __builtin_amdgcn_wmma_f32_16x16x32_f16(false, af, false, bf,
                                                 (short)0, acc, false, false);
  }
  float* dst = attnacc + (long)(b * 4 + hp) * 256;
#pragma unroll
  for (int r = 0; r < 8; ++r) {
    int M = r + 8 * kh;
    atomicAdd(&dst[M * 16 + col], acc[r]);
  }
}

// ---- norms + temperature + softmax, then fold proj: Meff = proj x attn ----
__global__ void CAB_sm_kernel(const float* __restrict__ attnacc,
                              const float* __restrict__ ssq_q,
                              const float* __restrict__ ssq_k,
                              const float* __restrict__ temp,
                              const float* __restrict__ proj,
                              float* __restrict__ meff) {
  const int b = blockIdx.x, tid = threadIdx.x;   // 64 threads
  __shared__ float nq[64], nk[64], sm[64][8];
  nq[tid] = fmaxf(sqrtf(ssq_q[b * 64 + tid]), 1e-12f);
  nk[tid] = fmaxf(sqrtf(ssq_k[b * 64 + tid]), 1e-12f);
  __syncthreads();
  {
    int h = tid >> 3, cp = tid & 7;
    int hp = h >> 1, sub = h & 1;
    const float* blk = attnacc + ((b * 4 + hp) * 16 + sub * 8 + cp) * 16 + sub * 8;
    float t = temp[h];
    float v[8], mx = -3.0e38f;
    for (int d = 0; d < 8; ++d) {
      v[d] = blk[d] / (nq[h * 8 + cp] * nk[h * 8 + d]) * t;
      mx = fmaxf(mx, v[d]);
    }
    float ssum = 0.0f;
    for (int d = 0; d < 8; ++d) { v[d] = __expf(v[d] - mx); ssum += v[d]; }
    for (int d = 0; d < 8; ++d) sm[tid][d] = v[d] / ssum;
  }
  __syncthreads();
  // Meff[b][co][h*8+d] = sum_cp proj[co][h*8+cp] * sm[h*8+cp][d]
  for (int h = 0; h < 8; ++h) {
    float acc[8] = {};
    for (int cp = 0; cp < 8; ++cp) {
      float pw = proj[tid * 64 + h * 8 + cp];
      for (int d = 0; d < 8; ++d) acc[d] += pw * sm[h * 8 + cp][d];
    }
    for (int d = 0; d < 8; ++d)
      meff[((long)b * 64 + tid) * 64 + h * 8 + d] = acc[d];
  }
}

// ----------------------------- host-side pipeline --------------------------
extern "C" void kernel_launch(void* const* d_in, const int* in_sizes, int n_in,
                              void* d_out, int out_size, void* d_ws, size_t ws_size,
                              hipStream_t stream) {
  const float* input1 = (const float*)d_in[0];
  const float* input2 = (const float*)d_in[1];
  const float* q_w    = (const float*)d_in[2];
  const float* q_dw   = (const float*)d_in[3];
  const float* kv_w   = (const float*)d_in[4];
  const float* kv_dw  = (const float*)d_in[5];
  const float* proj_w = (const float*)d_in[6];
  const float* temp   = (const float*)d_in[7];
  float* out = (float*)d_out;

  // workspace carve-up
  float*    t_q   = (float*)d_ws;                 // [4][64][65536]  f32
  float*    t_kv  = t_q + 16777216;               // [4][128][65536] f32
  _Float16* qhat  = (_Float16*)(t_kv + 33554432); // [4][64][65536]  f16
  _Float16* kvhat = qhat + 16777216;              // [4][128][65536] f16 (k|v)
  float*    smallb  = (float*)(kvhat + 33554432);
  float*    ssq_q   = smallb;                     // 256
  float*    ssq_k   = ssq_q + 256;                // 256
  float*    attnacc = ssq_k + 256;                // 4*4*256 = 4096
  float*    meff    = attnacc + 4096;             // 4*64*64 = 16384

  dim3 blk(256);
  CAB_zero_kernel<<<dim3((4608 + 255) / 256), blk, 0, stream>>>(smallb, 4608);

  // q = 1x1 conv (WMMA GEMM)
  CAB_gemm_kernel<float><<<dim3(1024, NB, 1), blk, 0, stream>>>(
      input1, q_w, t_q, 64L * NPIX, 64L * NPIX, 0);
  // kv = 1x1 conv, 128 outputs
  CAB_gemm_kernel<float><<<dim3(1024, NB, 2), blk, 0, stream>>>(
      input2, kv_w, t_kv, 64L * NPIX, 128L * NPIX, 0);

  // depthwise 3x3 + f16 cast + sum-of-squares (q fully; k = first 64 of kv)
  CAB_dw_kernel<<<dim3(256, 64, NB), blk, 0, stream>>>(t_q, q_dw, qhat, ssq_q, 64, 64);
  CAB_dw_kernel<<<dim3(256, 128, NB), blk, 0, stream>>>(t_kv, kv_dw, kvhat, ssq_k, 128, 64);

  // Gram matrices (q.k^T over pixels), WMMA + atomic reduction
  CAB_attn_kernel<<<dim3(4, 4, NB), blk, 0, stream>>>(qhat, kvhat, attnacc);

  // norms + softmax + fold projection into Meff
  CAB_sm_kernel<<<dim3(NB), dim3(64), 0, stream>>>(attnacc, ssq_q, ssq_k, temp,
                                                   proj_w, meff);

  // out = Meff @ v  (per-batch 64x64 weights, v = second half of kvhat)
  CAB_gemm_kernel<_Float16><<<dim3(1024, NB, 1), blk, 0, stream>>>(
      kvhat + 64L * NPIX, meff, out, 128L * NPIX, 64L * NPIX, 64 * 64);
}
